// GraphAttentionLayer_69501160784121
// MI455X (gfx1250) — matrix operand order repaired and checked by reading, not compiled, not run on hardware
//
#include <hip/hip_runtime.h>
#include <hip/hip_bf16.h>
#include <stdint.h>

#define GB 4
#define GN 4096
#define GF 128
#define GU 64
#define LEAKY 0.2f

typedef __attribute__((ext_vector_type(16))) _Float16 v16h;
typedef __attribute__((ext_vector_type(8)))  float    v8f;
typedef __attribute__((ext_vector_type(8)))  unsigned int v8u;
typedef __attribute__((ext_vector_type(4)))  unsigned int u32x4;
typedef __attribute__((ext_vector_type(8)))  int          i32x8;
typedef __attribute__((ext_vector_type(4)))  int          i32x4;

#if defined(__gfx1250__) && __has_builtin(__builtin_amdgcn_tensor_load_to_lds)
#define HAVE_TDM 1
#else
#define HAVE_TDM 0
#endif

// ---------------------------------------------------------------------------
// Kernel A: h = x @ W (fp32 VALU, exact), emit hT16[b][u][j] = (f16)h[b][j][u],
// and f1 = h@a1, f2 = h@a2 in fp32 via wave32 shuffle reductions.
// Grid: (GN/16, GB), 256 threads.
// ---------------------------------------------------------------------------
__global__ __launch_bounds__(256) void gat_h_kernel(
    const float* __restrict__ x, const float* __restrict__ W,
    const float* __restrict__ a, _Float16* __restrict__ hT16,
    float* __restrict__ f1g, float* __restrict__ f2g)
{
    __shared__ float Ws[GF * GU];     // 32 KB
    __shared__ float xs[16 * GF];     // 8 KB
    __shared__ float pf1[16][2];
    __shared__ float pf2[16][2];

    const int tid = threadIdx.x;
    const int b   = blockIdx.y;
    const int j0  = blockIdx.x * 16;

    // Stage W (128x64) and the 16 x-rows into LDS.
    {
        const float4* W4  = (const float4*)W;
        float4*       Ws4 = (float4*)Ws;
        for (int i = tid; i < (GF * GU) / 4; i += 256) Ws4[i] = W4[i];
        const float4* x4  = (const float4*)(x + ((size_t)b * GN + j0) * GF);
        float4*       xs4 = (float4*)xs;
        for (int i = tid; i < (16 * GF) / 4; i += 256) xs4[i] = x4[i];
    }
    __syncthreads();

    const int u    = tid & 63;   // output feature
    const int jj   = tid >> 6;   // 0..3 : row subgroup (uniform per wave-pair)
    const int lane = tid & 31;
    const float a1u = a[u];
    const float a2u = a[GU + u];

    float hv[4];
    #pragma unroll
    for (int rr = 0; rr < 4; ++rr) {
        const int row = jj + 4 * rr;   // 4 rows per thread
        float acc = 0.f;
        const float* xr = &xs[row * GF];
        #pragma unroll 8
        for (int f = 0; f < GF; ++f) acc = fmaf(xr[f], Ws[f * GU + u], acc);
        hv[rr] = acc;
        // transposed f16 copy for the WMMA B operand of the AV GEMM
        hT16[(size_t)(b * GU + u) * GN + (j0 + row)] = (_Float16)acc;
    }

    // f1/f2: reduce acc*a over u. Threads sharing jj span exactly 2 wave32s.
    #pragma unroll
    for (int rr = 0; rr < 4; ++rr) {
        const int row = jj + 4 * rr;
        float s1 = hv[rr] * a1u;
        float s2 = hv[rr] * a2u;
        #pragma unroll
        for (int off = 16; off > 0; off >>= 1) {
            s1 += __shfl_xor(s1, off, 32);
            s2 += __shfl_xor(s2, off, 32);
        }
        if (lane == 0) {
            const int half = u >> 5;   // which 32-lane half of the u range
            pf1[row][half] = s1;
            pf2[row][half] = s2;
        }
    }
    __syncthreads();
    if (tid < 16) {
        f1g[(size_t)b * GN + j0 + tid] = pf1[tid][0] + pf1[tid][1];
        f2g[(size_t)b * GN + j0 + tid] = pf2[tid][0] + pf2[tid][1];
    }
}

// ---------------------------------------------------------------------------
// TDM: DMA one 64(u) x 32(j) f16 tile of hT (row-major, ld=GN) into LDS.
// D# per CDNA5 ISA ch.8: 2D tensor, data_size=2B, groups 2/3 zero (<=2D).
// LDS result is row-major compacted: 64 rows x 64 bytes == hchunk layout.
// clang-23 toolchain: 6-arg builtin (g0, g1, g2, g3, extra, cpol).
// ---------------------------------------------------------------------------
#if HAVE_TDM
__device__ __forceinline__ void tdm_load_chunk(const _Float16* gsrc,
                                               unsigned int lds_byte_addr)
{
    const unsigned long long ga = (unsigned long long)(uintptr_t)gsrc;
    u32x4 g0;
    g0[0] = 1u;                                    // count=1 (valid), user mode
    g0[1] = lds_byte_addr;                         // lds_addr [63:32]
    g0[2] = (unsigned int)ga;                      // global_addr[31:0]
    g0[3] = ((unsigned int)(ga >> 32) & 0x1FFFFFFu) | (2u << 30); // [56:32]|type=2

    i32x8 g1;
    g1[0] = 0x00010000;          // workgroup_mask=0, data_size=1 (2 bytes)
    g1[1] = (int)(0x1000u << 16);// atomic_barrier=0 | tensor_dim0[15:0]=4096
    g1[2] = (int)((unsigned)GU << 16); // tensor_dim0[31:16]=0 | tensor_dim1[15:0]=64
    g1[3] = (int)(32u << 16);    // tensor_dim1[31:16]=0 | tile_dim0=32
    g1[4] = GU;                  // tile_dim1=64 | tile_dim2=0
    g1[5] = GN;                  // tensor_dim0_stride[31:0]=4096
    g1[6] = 0;                   // stride[47:32]=0 | tensor_dim1_stride lo=0
    g1[7] = 0;

    i32x4 gz4 = {0, 0, 0, 0};
    i32x8 gz8 = {0, 0, 0, 0, 0, 0, 0, 0};
    __builtin_amdgcn_tensor_load_to_lds(g0, g1, gz4, gz4, gz8, 0);
}
#endif

// ---------------------------------------------------------------------------
// Kernel B: flash-style fused softmax(leakyrelu(f1_i + f2_j)) @ h using
// V_WMMA_F32_16X16X32_F16.  Each wave32 owns a 16-row block (16x64 output,
// 4 v8f accumulators); 4 waves per workgroup share the LDS h-chunk, which is
// double-buffered and filled by the Tensor Data Mover (TENSORcnt-tracked),
// overlapping DMA of chunk c+1 with WMMA+exp of chunk c.
// Grid: (GN/64, GB), 128 threads.
// ---------------------------------------------------------------------------
__global__ __launch_bounds__(128) void gat_attn_kernel(
    const _Float16* __restrict__ hT16, const float* __restrict__ f1g,
    const float* __restrict__ f2g, float* __restrict__ out)
{
    __shared__ float        f2s[GN];                // 16 KB: whole f2 row
    __shared__ unsigned int hchunk[2 * GU * 16];    // 2 x 4 KB double buffer
    __shared__ float        wmax[4];

    const int tid  = threadIdx.x;
    const int b    = blockIdx.y;
    const int lane = tid & 31;
    const int wave = tid >> 5;
    const int i0   = (blockIdx.x * 4 + wave) * 16;   // this wave's 16 output rows

    // ---- Stage f2[b][*] into LDS and reduce its max (for stable softmax) ----
    float lmax = -3.4e38f;
    {
        const float4* f2v  = (const float4*)(f2g + (size_t)b * GN);
        float4*       f2sv = (float4*)f2s;
        for (int i = tid; i < GN / 4; i += 128) {
            float4 v = f2v[i];
            f2sv[i] = v;
            lmax = fmaxf(lmax, fmaxf(fmaxf(v.x, v.y), fmaxf(v.z, v.w)));
        }
    }
    #pragma unroll
    for (int off = 16; off > 0; off >>= 1)
        lmax = fmaxf(lmax, __shfl_xor(lmax, off, 32));
    if (lane == 0) wmax[wave] = lmax;
    __syncthreads();
    const float maxf2 = fmaxf(fmaxf(wmax[0], wmax[1]), fmaxf(wmax[2], wmax[3]));

    const int m   = lane & 15;   // row within wave's block / B column n
    const int khi = lane >> 4;   // which K half this lane covers
    const float f1m = f1g[(size_t)b * GN + i0 + m];
    float emax = f1m + maxf2;                       // leakyrelu is monotone:
    emax = (emax >= 0.f) ? emax : LEAKY * emax;     // rowmax = lrelu(f1+max f2)

    v8f acc[4] = {};      // 16x64 fp32 accumulators
    float rowsum = 0.f;   // softmax denominator partial for row m

    const _Float16* hTb = hT16 + (size_t)b * GU * GN;
#if HAVE_TDM
    const unsigned int lds_hchunk = (unsigned int)(uintptr_t)&hchunk[0];
    if (wave == 0) tdm_load_chunk(hTb, lds_hchunk);           // preload chunk 0
#else
    const uint4* hT4 = (const uint4*)hTb;
#endif

    for (int c = 0; c < GN / 32; ++c) {
        const int j0 = c * 32;
#if HAVE_TDM
        const unsigned int* hcur = hchunk + (c & 1) * (GU * 16);
        if (wave == 0) __builtin_amdgcn_s_wait_tensorcnt(0);  // chunk c landed
        __syncthreads();              // chunk c visible; chunk c-1 fully consumed
        if (wave == 0 && c + 1 < GN / 32)                     // overlap: DMA c+1
            tdm_load_chunk(hTb + (j0 + 32),
                           lds_hchunk + (unsigned)(((c + 1) & 1) * (GU * 16 * 4)));
#else
        const unsigned int* hcur = hchunk;
        __syncthreads();   // previous chunk fully consumed
        for (int idx = tid; idx < 256; idx += 128) {
            const int uu = idx >> 2;
            const int q  = idx & 3;
            ((uint4*)hchunk)[idx] = hT4[(uu * GN + j0) / 8 + q];
        }
        __syncthreads();
#endif

        // ---- A operand: 16x32 f16 tile of probabilities, built directly in
        //      contiguous K order (lane<16: K in {0..15}; lane>=16: +16).
        union { _Float16 f[16]; v16h v; } A;
        #pragma unroll
        for (int vi = 0; vi < 8; ++vi) {
            const int k0 = (2 * vi) + (khi << 4);
            float e0 = f1m + f2s[j0 + k0];
            e0 = (e0 >= 0.f) ? e0 : LEAKY * e0;
            float e1 = f1m + f2s[j0 + k0 + 1];
            e1 = (e1 >= 0.f) ? e1 : LEAKY * e1;
            const float p0 = __expf(e0 - emax);
            const float p1 = __expf(e1 - emax);
            rowsum += p0 + p1;
            A.f[2 * vi]     = (_Float16)p0;
            A.f[2 * vi + 1] = (_Float16)p1;
        }

        // ---- B operands + WMMA: 4 column tiles of 16 u's each.
        //      B VGPR v = packed {h[K],h[K+1]} at column n=m, K = 2v (+16 for
        //      lanes>=16) -> one contiguous dword per VGPR from LDS.
        #pragma unroll
        for (int t = 0; t < 4; ++t) {
            const unsigned int* brow = &hcur[(16 * t + m) * 16 + (khi << 3)];
            v8u braw;
            #pragma unroll
            for (int vi = 0; vi < 8; ++vi) braw[vi] = brow[vi];
            const v16h bv = __builtin_bit_cast(v16h, braw);
            acc[t] = __builtin_amdgcn_wmma_f32_16x16x32_f16(
                false, A.v, false, bv, (short)0, acc[t], false, false);
        }
    }

    // ---- Softmax normalization + store. Lanes l and l+16 hold partial sums
    //      for the same row m -> pairwise add, then per-VGPR broadcast.
    const float Z = rowsum + __shfl_xor(rowsum, 16, 32);   // full denom of row m
    #pragma unroll
    for (int r = 0; r < 8; ++r) {
        const int rowm = r + (khi << 3);                   // C/D layout row
        const float zr  = __shfl(Z, rowm, 32);             // Z[rowm] lives on lane rowm
        const float inv = 1.0f / zr;
        float* orow = out + ((size_t)b * GN + i0 + rowm) * GU;
        #pragma unroll
        for (int t = 0; t < 4; ++t)
            orow[16 * t + m] = acc[t][r] * inv;
    }
}

// ---------------------------------------------------------------------------
extern "C" void kernel_launch(void* const* d_in, const int* in_sizes, int n_in,
                              void* d_out, int out_size, void* d_ws, size_t ws_size,
                              hipStream_t stream) {
    const float* x = (const float*)d_in[0];   // (4, 4096, 128)
    const float* W = (const float*)d_in[1];   // (128, 64)
    const float* a = (const float*)d_in[2];   // (128, 1)
    float* outp = (float*)d_out;              // (4, 4096, 64)

    // Workspace layout: hT16 (b,u,j) f16 | f1 (b,n) f32 | f2 (b,n) f32
    _Float16* hT16 = (_Float16*)d_ws;
    float* f1 = (float*)((char*)d_ws + (size_t)GB * GU * GN * sizeof(_Float16));
    float* f2 = f1 + (size_t)GB * GN;

    dim3 gridA(GN / 16, GB);
    gat_h_kernel<<<gridA, 256, 0, stream>>>(x, W, a, hT16, f1, f2);

    dim3 gridB(GN / 64, GB);
    gat_attn_kernel<<<gridB, 128, 0, stream>>>(hT16, f1, f2, outp);
}
